// Net_13176959664345
// MI455X (gfx1250) — matrix operand order repaired and checked by reading, not compiled, not run on hardware
//
#include <hip/hip_runtime.h>
#include <hip/hip_bf16.h>
#include <math.h>
#include <stdint.h>

// ---------------------------------------------------------------------------
// PointNet++ (pointnet2 segmentation) forward for MI455X / gfx1250.
// All linear layers run through a wave32 WMMA f16->f32 GEMM
// (v_wmma_f32_16x16x32_f16). Activations staged as f16 with K padded to
// multiples of 32; weights pre-transposed to [N, Kpad] f16 so both A and B
// fragments are contiguous vector loads.
// ---------------------------------------------------------------------------

typedef __attribute__((ext_vector_type(16))) _Float16 v16h;
typedef __attribute__((ext_vector_type(8)))  _Float16 v8h;
typedef __attribute__((ext_vector_type(8)))  float    v8f;

#define EW_BLOCK 256

// ---------------- WMMA GEMM: C[M,N] = A[M,Kpad](f16) * Bt[N,Kpad]^T + bias --
// One wave per 16x16 output tile. A-fragment k order used here (contiguous, not the ISA table order):
// lane L (r=L%16, hi=L>=16): a[i]=A[r, hi*16+i] (same k order as B).
// B stored transposed [N,Kpad]: lane L holds col n=r, k = hi*16 + i.
__global__ void __launch_bounds__(256)
wmma_gemm_kernel(const _Float16* __restrict__ A, const _Float16* __restrict__ Bt,
                 const float* __restrict__ bias, float* __restrict__ C,
                 int Mt, int Nt, int Kpad, int relu) {
    int wave = threadIdx.x >> 5;
    int lane = threadIdx.x & 31;
    int tile = blockIdx.x * 8 + wave;          // wave-uniform
    if (tile >= Mt * Nt) return;               // whole wave exits together
    int tm = tile / Nt, tn = tile % Nt;
    int r  = lane & 15;
    int hi = lane >> 4;                        // 0 or 1

    const _Float16* arow = A  + (size_t)(tm * 16 + r) * Kpad + hi * 16;
    const _Float16* brow = Bt + (size_t)(tn * 16 + r) * Kpad + hi * 16;

    v8f acc = {};
    for (int k0 = 0; k0 < Kpad; k0 += 32) {
        if (k0 + 32 < Kpad) {                  // gfx1250 global_prefetch_b8
            __builtin_prefetch(arow + k0 + 32, 0, 0);
            __builtin_prefetch(brow + k0 + 32, 0, 0);
        }
        v8h a0 = *(const v8h*)(arow + k0);          // K = sel16 + 0..7
        v8h a1 = *(const v8h*)(arow + k0 + 8);      // K = sel16 + 8..15
        v16h av, bv;
#pragma unroll
        for (int i = 0; i < 8; ++i) { av[i] = a0[i]; av[i + 8] = a1[i]; }
        bv = *(const v16h*)(brow + k0);             // K = sel16 + 0..15
        acc = __builtin_amdgcn_wmma_f32_16x16x32_f16(
            /*neg_a=*/false, av, /*neg_b=*/false, bv,
            /*c_mod=*/(short)0, acc, /*reuse_a=*/false, /*reuse_b=*/false);
    }

    int n = tn * 16 + r;
    int N = Nt * 16;
    float bvs = bias ? bias[n] : 0.f;
    int mbase = tm * 16 + hi * 8;
#pragma unroll
    for (int v = 0; v < 8; ++v) {
        float val = acc[v] + bvs;
        if (relu) val = fmaxf(val, 0.f);
        C[(size_t)(mbase + v) * N + n] = val;
    }
}

// ---------------- Farthest point sampling (single block, serial scan) ------
__global__ void __launch_bounds__(1024)
fps_kernel(const float* __restrict__ pos, int n, int m, int* __restrict__ sidx) {
    __shared__ float sb[1024];
    __shared__ int   si[1024];
    __shared__ float slast[3];
    __shared__ int   slidx;
    const int tid = threadIdx.x;
    const int MAXPT = 16;
    float px[MAXPT], py[MAXPT], pz[MAXPT], md[MAXPT];
    int cnt = 0;
    for (int k = 0; k < MAXPT; ++k) {
        int i = tid + k * 1024;
        if (i < n) {
            px[k] = pos[i * 3]; py[k] = pos[i * 3 + 1]; pz[k] = pos[i * 3 + 2];
            md[k] = 3.4e38f; cnt = k + 1;
        }
    }
    if (tid == 0) slidx = 0;
    __syncthreads();
    for (int it = 0; it < m; ++it) {
        int last = slidx;
        if (tid == 0) {
            sidx[it] = last;
            slast[0] = pos[last * 3]; slast[1] = pos[last * 3 + 1]; slast[2] = pos[last * 3 + 2];
        }
        __syncthreads();
        float lx = slast[0], ly = slast[1], lz = slast[2];
        float best = -1.f; int bidx = 0x7fffffff;
        for (int k = 0; k < cnt; ++k) {
            float dx = px[k] - lx, dy = py[k] - ly, dz = pz[k] - lz;
            float d = dx * dx + dy * dy + dz * dz;
            md[k] = fminf(md[k], d);
            if (md[k] > best) { best = md[k]; bidx = tid + k * 1024; }
        }
        sb[tid] = best; si[tid] = bidx;
        __syncthreads();
        for (int s = 512; s > 0; s >>= 1) {
            if (tid < s) {
                float ob = sb[tid + s]; int oi = si[tid + s];
                if (ob > sb[tid] || (ob == sb[tid] && oi < si[tid])) { sb[tid] = ob; si[tid] = oi; }
            }
            __syncthreads();
        }
        if (tid == 0) slidx = si[0];
        __syncthreads();
    }
}

// ---------------- K nearest neighbors (brute force, thread per query) ------
__global__ void knn_kernel(const float* __restrict__ src, int nsrc,
                           const float* __restrict__ dst, int ndst,
                           int K, int* __restrict__ nidx, float* __restrict__ nd2) {
    int i = blockIdx.x * blockDim.x + threadIdx.x;
    if (i >= ndst) return;
    float cx = dst[i * 3], cy = dst[i * 3 + 1], cz = dst[i * 3 + 2];
    float bd[64]; int bi[64];
    int cnt = 0, wpos = 0; float worst = -1.f;
    for (int j = 0; j < nsrc; ++j) {
        float dx = src[j * 3] - cx, dy = src[j * 3 + 1] - cy, dz = src[j * 3 + 2] - cz;
        float d = dx * dx + dy * dy + dz * dz;
        if (cnt < K) {
            bd[cnt] = d; bi[cnt] = j;
            if (d > worst) { worst = d; wpos = cnt; }
            cnt++;
        } else if (d < worst) {
            bd[wpos] = d; bi[wpos] = j;
            worst = bd[0]; wpos = 0;
            for (int t = 1; t < K; ++t) if (bd[t] > worst) { worst = bd[t]; wpos = t; }
        }
    }
    for (int t = 0; t < K; ++t) {
        nidx[(size_t)i * K + t] = bi[t];
        nd2 [(size_t)i * K + t] = bd[t];
    }
}

// ---------------- helpers: gathers, conversions, concat builders -----------
__global__ void gather3_kernel(const float* __restrict__ pos, const int* __restrict__ sidx,
                               int m, int mpad, float* __restrict__ ctr) {
    int t = blockIdx.x * blockDim.x + threadIdx.x;
    if (t >= mpad * 3) return;
    int i = t / 3, d = t % 3;
    ctr[t] = (i < m) ? pos[(size_t)sidx[i] * 3 + d] : 0.f;
}

__global__ void wcvt_kernel(const float* __restrict__ w, int K, int N, int Kpad,
                            _Float16* __restrict__ wt) {
    int t = blockIdx.x * blockDim.x + threadIdx.x;
    if (t >= N * Kpad) return;
    int n = t / Kpad, kp = t % Kpad;
    wt[t] = (_Float16)((kp < K) ? w[(size_t)kp * N + n] : 0.f);   // transpose [K,N]->[N,Kpad]
}

__global__ void cvt_f16_kernel(const float* __restrict__ x, int M, int C,
                               int Mpad, int Cpad, _Float16* __restrict__ A) {
    int t = blockIdx.x * blockDim.x + threadIdx.x;
    if (t >= Mpad * Cpad) return;
    int rw = t / Cpad, c = t % Cpad;
    float v = (rw < M && c < C) ? x[(size_t)rw * C + c] : 0.f;
    A[t] = (_Float16)v;
}

// SA message input: [x_j , pos_j - ctr_i] for each (center i, neighbor j)
__global__ void build_sa_in_kernel(const float* __restrict__ x, int C,
                                   const float* __restrict__ psrc, const float* __restrict__ ctr,
                                   const int* __restrict__ nidx, int m, int K, int Cpad,
                                   _Float16* __restrict__ A) {
    int t = blockIdx.x * blockDim.x + threadIdx.x;
    int total = m * K * Cpad;
    if (t >= total) return;
    int row = t / Cpad, c = t % Cpad;
    int i = row / K;
    int j = nidx[row];
    float v = 0.f;
    if (c < C) v = x[(size_t)j * C + c];
    else if (c < C + 3) { int d = c - C; v = psrc[(size_t)j * 3 + d] - ctr[(size_t)i * 3 + d]; }
    A[t] = (_Float16)v;
}

__global__ void concat2_f16_kernel(const float* __restrict__ a, int Ca, int abcast,
                                   const float* __restrict__ b, int Cb,
                                   int M, int Mpad, int Cpad, _Float16* __restrict__ A) {
    int t = blockIdx.x * blockDim.x + threadIdx.x;
    if (t >= Mpad * Cpad) return;
    int rw = t / Cpad, c = t % Cpad;
    float v = 0.f;
    if (rw < M) {
        if (c < Ca)            v = a[(size_t)(abcast ? 0 : rw) * Ca + c];
        else if (c < Ca + Cb)  v = b[(size_t)rw * Cb + (c - Ca)];
    }
    A[t] = (_Float16)v;
}

// knn_interpolate (1/d^2 weights) fused with skip concat
__global__ void interp_concat_f16_kernel(const float* __restrict__ xsrc, int C,
                                         const int* __restrict__ nidx, const float* __restrict__ nd2,
                                         int k, const float* __restrict__ xskip, int Cskip,
                                         int M, int Mpad, int Cpad, _Float16* __restrict__ A) {
    int t = blockIdx.x * blockDim.x + threadIdx.x;
    if (t >= Mpad * Cpad) return;
    int row = t / Cpad, c = t % Cpad;
    float v = 0.f;
    if (row < M) {
        if (c < C) {
            float ws = 0.f, acc = 0.f;
            for (int j = 0; j < k; ++j) {
                float d2 = nd2[(size_t)row * k + j];
                float w  = 1.f / fmaxf(d2, 1e-16f);
                acc += w * xsrc[(size_t)nidx[(size_t)row * k + j] * C + c];
                ws  += w;
            }
            v = acc / ws;
        } else if (c < C + Cskip) {
            v = xskip[(size_t)row * Cskip + (c - C)];
        }
    }
    A[t] = (_Float16)v;
}

// ---------------- BatchNorm (training batch stats) -------------------------
__global__ void bn_stats_kernel(const float* __restrict__ x, int M, int N,
                                float* __restrict__ mu, float* __restrict__ rsig) {
    __shared__ float s1[256], s2[256];
    int n = blockIdx.x, tid = threadIdx.x;
    float a = 0.f, b = 0.f;
    for (int r = tid; r < M; r += 256) {
        float v = x[(size_t)r * N + n];
        a += v; b += v * v;
    }
    s1[tid] = a; s2[tid] = b;
    __syncthreads();
    for (int s = 128; s > 0; s >>= 1) {
        if (tid < s) { s1[tid] += s1[tid + s]; s2[tid] += s2[tid + s]; }
        __syncthreads();
    }
    if (tid == 0) {
        float m_  = s1[0] / (float)M;
        float var = s2[0] / (float)M - m_ * m_;
        mu[n]   = m_;
        rsig[n] = rsqrtf(fmaxf(var, 0.f) + 1e-5f);
    }
}

__global__ void bn_apply_relu_kernel(float* __restrict__ x, int total, int N,
                                     const float* __restrict__ mu, const float* __restrict__ rsig,
                                     const float* __restrict__ g, const float* __restrict__ beta) {
    int t = blockIdx.x * blockDim.x + threadIdx.x;
    if (t >= total) return;
    int n = t % N;
    float v = x[t];
    v = g[n] * (v - mu[n]) * rsig[n] + beta[n];
    x[t] = fmaxf(v, 0.f);
}

// ---------------- pooling / output -----------------------------------------
__global__ void sa_maxpool_kernel(const float* __restrict__ h, const float* __restrict__ d2,
                                  float r2, int m, int K, int N, float* __restrict__ out) {
    int t = blockIdx.x * blockDim.x + threadIdx.x;
    if (t >= m * N) return;
    int i = t / N, n = t % N;
    float best = -3.4e38f; int found = 0;
    for (int j = 0; j < K; ++j) {
        if (d2[(size_t)i * K + j] <= r2) {
            best = fmaxf(best, h[(size_t)(i * K + j) * N + n]);
            found = 1;
        }
    }
    out[(size_t)i * N + n] = found ? best : 0.f;
}

__global__ void gmaxpool_kernel(const float* __restrict__ g, int M, int N,
                                float* __restrict__ out) {
    int n = blockIdx.x * blockDim.x + threadIdx.x;
    if (n >= N) return;
    float b = -3.4e38f;
    for (int r = 0; r < M; ++r) b = fmaxf(b, g[(size_t)r * N + n]);
    out[n] = b;
}

__global__ void log_softmax16_kernel(const float* __restrict__ x, int M,
                                     float* __restrict__ out) {
    int i = blockIdx.x * blockDim.x + threadIdx.x;
    if (i >= M) return;
    float mx = -3.4e38f;
    for (int c = 0; c < 16; ++c) mx = fmaxf(mx, x[i * 16 + c]);
    float s = 0.f;
    for (int c = 0; c < 16; ++c) s += expf(x[i * 16 + c] - mx);
    float l = logf(s);
    for (int c = 0; c < 16; ++c) out[i * 16 + c] = x[i * 16 + c] - mx - l;
}

// ---------------------------------------------------------------------------
// Host orchestration
// ---------------------------------------------------------------------------
extern "C" void kernel_launch(void* const* d_in, const int* in_sizes, int n_in,
                              void* d_out, int out_size, void* d_ws, size_t ws_size,
                              hipStream_t stream) {
    (void)in_sizes; (void)n_in; (void)out_size; (void)ws_size;

    // Sizes
    const int N0 = 16384;
    const int M1 = 3277, M1P = 3280;   // ceil(16384*0.2)
    const int M2 = 820,  M2P = 832;    // ceil(3277*0.25)
    const int KN = 64;
    const int R1 = M1 * KN;            // 209728 (mult of 16)
    const int R2 = M2 * KN;            // 52480  (mult of 16)

    // 19 linear layers, flattened in setup_inputs() insertion order:
    // pos, then sa1,sa2,sa3,fp3,fp2,fp1,head; each layer = {w,b,g,beta}.
    const int NL = 19;
    static const int LK[NL] = {6,64,64, 131,128,128, 259,256,512, 1280,256, 384,256, 131,128,128, 128,128,128};
    static const int LN[NL] = {64,64,128, 128,128,256, 256,512,1024, 256,256, 256,128, 128,128,128, 128,128,16};

    const float* pos = (const float*)d_in[0];
    const float* Wf[NL]; const float* Bf[NL]; const float* Gf[NL]; const float* BTf[NL];
    for (int i = 0; i < NL; ++i) {
        Wf[i]  = (const float*)d_in[1 + 4 * i];
        Bf[i]  = (const float*)d_in[2 + 4 * i];
        Gf[i]  = (const float*)d_in[3 + 4 * i];
        BTf[i] = (const float*)d_in[4 + 4 * i];
    }

    // Workspace bump allocator (256B aligned)
    uintptr_t base = (uintptr_t)d_ws;
    size_t off = 0;
    auto alloc = [&](size_t bytes) -> void* {
        off = (off + 255) & ~(size_t)255;
        void* p = (void*)(base + off);
        off += bytes;
        return p;
    };
    auto pad32 = [](int k) { return (k + 31) & ~31; };
    auto gs = [](size_t n) { return dim3((unsigned)((n + EW_BLOCK - 1) / EW_BLOCK)); };

    _Float16* WT16[NL]; int KP[NL];
    for (int i = 0; i < NL; ++i) {
        KP[i] = pad32(LK[i]);
        WT16[i] = (_Float16*)alloc((size_t)LN[i] * KP[i] * sizeof(_Float16));
    }
    float*    PING = (float*)alloc((size_t)R1 * 128 * sizeof(float));
    float*    PONG = (float*)alloc((size_t)R1 * 128 * sizeof(float));
    _Float16* AST  = (_Float16*)alloc((size_t)R1 * 64 * sizeof(_Float16));
    int*   nidx  = (int*)alloc((size_t)R1 * sizeof(int));
    float* nd2   = (float*)alloc((size_t)R1 * sizeof(float));
    int*   sidx1 = (int*)alloc(M1 * sizeof(int));
    int*   sidx2 = (int*)alloc(M2 * sizeof(int));
    float* ctr1  = (float*)alloc((size_t)M1P * 3 * sizeof(float));
    float* ctr2  = (float*)alloc((size_t)M2P * 3 * sizeof(float));
    float* x1    = (float*)alloc((size_t)M1P * 128 * sizeof(float));
    float* x2    = (float*)alloc((size_t)M2P * 256 * sizeof(float));
    float* x3    = (float*)alloc(1024 * sizeof(float));
    float* y2    = (float*)alloc((size_t)M2P * 256 * sizeof(float));
    float* y1    = (float*)alloc((size_t)M1P * 128 * sizeof(float));
    float* y0    = (float*)alloc((size_t)N0 * 128 * sizeof(float));
    float* mu    = (float*)alloc(1024 * sizeof(float));
    float* rsig  = (float*)alloc(1024 * sizeof(float));

    // Convert + transpose weights once per launch (cheap, ~1.4M elements)
    for (int i = 0; i < NL; ++i)
        wcvt_kernel<<<gs((size_t)LN[i] * KP[i]), EW_BLOCK, 0, stream>>>(Wf[i], LK[i], LN[i], KP[i], WT16[i]);

    auto gemm = [&](const _Float16* A, int li, int Mpad, float* C, int relu) {
        int Mt = Mpad / 16, Nt = LN[li] / 16;
        int tiles = Mt * Nt;
        wmma_gemm_kernel<<<dim3((unsigned)((tiles + 7) / 8)), 256, 0, stream>>>(
            A, WT16[li], Bf[li], C, Mt, Nt, KP[li], relu);
    };

    // Run an MLP stage: first-layer A already staged in AST
    auto mlp = [&](int l0, int nl, int M, int Mpad, bool norm, float* finalOut) {
        const _Float16* Acur = AST;
        for (int i = 0; i < nl; ++i) {
            int li = l0 + i;
            bool last = (i == nl - 1);
            float* Cout = last ? finalOut : ((i & 1) ? PONG : PING);
            gemm(Acur, li, Mpad, Cout, (!norm && !last) ? 1 : 0);
            if (!last) {
                int Nn = LN[li];
                if (norm) {
                    bn_stats_kernel<<<dim3((unsigned)Nn), 256, 0, stream>>>(Cout, M, Nn, mu, rsig);
                    bn_apply_relu_kernel<<<gs((size_t)Mpad * Nn), EW_BLOCK, 0, stream>>>(
                        Cout, Mpad * Nn, Nn, mu, rsig, Gf[li], BTf[li]);
                }
                cvt_f16_kernel<<<gs((size_t)Mpad * KP[li + 1]), EW_BLOCK, 0, stream>>>(
                    Cout, M, Nn, Mpad, KP[li + 1], AST);
                Acur = AST;
            }
        }
    };

    // ---------------- SA1: FPS -> KNN(64) -> MLP[6,64,64,128] -> radius max
    fps_kernel<<<1, 1024, 0, stream>>>(pos, N0, M1, sidx1);
    gather3_kernel<<<gs((size_t)M1P * 3), EW_BLOCK, 0, stream>>>(pos, sidx1, M1, M1P, ctr1);
    knn_kernel<<<gs(M1), EW_BLOCK, 0, stream>>>(pos, N0, ctr1, M1, KN, nidx, nd2);
    build_sa_in_kernel<<<gs((size_t)R1 * 32), EW_BLOCK, 0, stream>>>(pos, 3, pos, ctr1, nidx, M1, KN, 32, AST);
    mlp(0, 3, R1, R1, true, PING);
    sa_maxpool_kernel<<<gs((size_t)M1 * 128), EW_BLOCK, 0, stream>>>(PING, nd2, 0.04f, M1, KN, 128, x1);

    // ---------------- SA2: FPS -> KNN(64) -> MLP[131,128,128,256] -> radius max
    fps_kernel<<<1, 1024, 0, stream>>>(ctr1, M1, M2, sidx2);
    gather3_kernel<<<gs((size_t)M2P * 3), EW_BLOCK, 0, stream>>>(ctr1, sidx2, M2, M2P, ctr2);
    knn_kernel<<<gs(M2), EW_BLOCK, 0, stream>>>(ctr1, M1, ctr2, M2, KN, nidx, nd2);
    build_sa_in_kernel<<<gs((size_t)R2 * 160), EW_BLOCK, 0, stream>>>(x1, 128, ctr1, ctr2, nidx, M2, KN, 160, AST);
    mlp(3, 3, R2, R2, true, PING);
    sa_maxpool_kernel<<<gs((size_t)M2 * 256), EW_BLOCK, 0, stream>>>(PING, nd2, 0.16f, M2, KN, 256, x2);

    // ---------------- SA3: MLP[259,256,512,1024] -> global max
    concat2_f16_kernel<<<gs((size_t)M2P * 288), EW_BLOCK, 0, stream>>>(x2, 256, 0, ctr2, 3, M2, M2P, 288, AST);
    mlp(6, 3, M2, M2P, true, PING);
    gmaxpool_kernel<<<gs(1024), EW_BLOCK, 0, stream>>>(PING, M2, 1024, x3);

    // ---------------- FP3: broadcast x3, concat x2, MLP[1280,256,256]
    concat2_f16_kernel<<<gs((size_t)M2P * 1280), EW_BLOCK, 0, stream>>>(x3, 1024, 1, x2, 256, M2, M2P, 1280, AST);
    mlp(9, 2, M2, M2P, true, y2);

    // ---------------- FP2: knn(k=3) interp y2 -> pos1, concat x1, MLP[384,256,128]
    knn_kernel<<<gs(M1), EW_BLOCK, 0, stream>>>(ctr2, M2, ctr1, M1, 3, nidx, nd2);
    interp_concat_f16_kernel<<<gs((size_t)M1P * 384), EW_BLOCK, 0, stream>>>(
        y2, 256, nidx, nd2, 3, x1, 128, M1, M1P, 384, AST);
    mlp(11, 2, M1, M1P, true, y1);

    // ---------------- FP1: knn(k=3) interp y1 -> pos, concat pos, MLP[131,128,128,128]
    knn_kernel<<<gs(N0), EW_BLOCK, 0, stream>>>(ctr1, M1, pos, N0, 3, nidx, nd2);
    interp_concat_f16_kernel<<<gs((size_t)N0 * 160), EW_BLOCK, 0, stream>>>(
        y1, 128, nidx, nd2, 3, pos, 3, N0, N0, 160, AST);
    mlp(13, 3, N0, N0, true, y0);

    // ---------------- head: MLP[128,128,128,16] (no BN, ReLU on hidden) + log_softmax
    cvt_f16_kernel<<<gs((size_t)N0 * 128), EW_BLOCK, 0, stream>>>(y0, N0, 128, N0, 128, AST);
    mlp(16, 3, N0, N0, false, PING);
    log_softmax16_kernel<<<gs(N0), EW_BLOCK, 0, stream>>>(PING, N0, (float*)d_out);
}
